// NLPConstraintVector_81329500717473
// MI455X (gfx1250) — matrix operand order repaired and checked by reading, not compiled, not run on hardware
//
#include <hip/hip_runtime.h>
#include <hip/hip_bf16.h>
#include <cstdint>
#include <cmath>

// ---------------------------------------------------------------------------
// Types for CDNA5 WMMA (wave32, 16x16x32 bf16 -> f32)
// ---------------------------------------------------------------------------
typedef __attribute__((ext_vector_type(16))) __bf16 v16bf;
typedef __attribute__((ext_vector_type(8)))  float  v8f;

union FragBF {
    v16bf    v;
    uint32_t d[8];
};

#define BERT_DIM 768
#define BATCH    256
#define SEQ      512
#define N_ROWS   5
#define N_COLS   12

// ---------------------------------------------------------------------------
// fp32 -> bf16 (round to nearest even) helpers
// ---------------------------------------------------------------------------
__device__ __forceinline__ uint16_t f2bf(float x) {
    uint32_t u = __builtin_bit_cast(uint32_t, x);
    uint32_t r = u + 0x7FFFu + ((u >> 16) & 1u);
    return (uint16_t)(r >> 16);
}
__device__ __forceinline__ uint32_t pack2(float a, float b) {
    return (uint32_t)f2bf(a) | ((uint32_t)f2bf(b) << 16);
}

// ---------------------------------------------------------------------------
// A fragment per ISA 16-bit A-matrix 16x32 layout (CDNA5 ISA 7.12.2):
//   lanes 0-15 : row M = lane,    K = {0..7} (dw 0..3) and {16..23} (dw 8..11)
//   lanes 16-31: row M = lane-16, K = {8..15} (dw 4..7) and {24..31} (dw 12..15)
// tileBase points at (row0, k0) in a packed-bf16 (dword) array.
// ---------------------------------------------------------------------------
__device__ __forceinline__ v16bf load_frag_a(const uint32_t* tileBase,
                                             int strideDw, int lane) {
    int row  = lane & 15;
    int half = (lane >> 4) & 1;
    const uint32_t* p = tileBase + row * strideDw + half * 4;
    FragBF f;
#pragma unroll
    for (int i = 0; i < 4; ++i) {
        f.d[i]     = p[i];
        f.d[4 + i] = p[8 + i];
    }
    return f.v;
}

// ---------------------------------------------------------------------------
// B fragment per ISA 16-bit B-matrix 32x16 layout (mirrors sparse-B table):
//   lanes 0-15 : col N = lane,    K = {0..7} and {16..23}
//   lanes 16-31: col N = lane-16, K = {8..15} and {24..31}
// Weights stored row-major [outDim][768] bf16 (384 dwords/row); lane reads two
// runs of 4 contiguous dwords, 8 apart, from its output-unit row.
// ---------------------------------------------------------------------------
__device__ __forceinline__ v16bf load_frag_b(const uint32_t* wbf,
                                             int n0, int k0dw, int lane) {
    int col  = n0 + (lane & 15);
    int half = (lane >> 4) & 1;
    const uint32_t* p = wbf + col * 384 + k0dw + half * 4;
    FragBF f;
#pragma unroll
    for (int i = 0; i < 4; ++i) {
        f.d[i]     = p[i];
        f.d[4 + i] = p[8 + i];
    }
    return f.v;
}

// ===========================================================================
// Kernel 1: convert att_w1 [128,768] and proj_w1 [256,768] to bf16 workspace
// ===========================================================================
__global__ __launch_bounds__(256) void cvt_weights_kernel(
    const float* __restrict__ att_w1, const float* __restrict__ proj_w1,
    uint16_t* __restrict__ w1bf, uint16_t* __restrict__ pw1bf) {
    int i = blockIdx.x * 256 + threadIdx.x;
    if (i < 128 * BERT_DIM) w1bf[i] = f2bf(att_w1[i]);
    if (i < 256 * BERT_DIM) pw1bf[i] = f2bf(proj_w1[i]);
}

// ===========================================================================
// Kernel 2: scores[b,s] = att_w2 . tanh(token_hidden @ att_w1^T + b1) + b2,
//           masked. 128 rows x 128 cols per block, K=768, bf16 WMMA.
//           HBM-bound: streams 402 MB of fp32 activations once.
// ===========================================================================
__global__ __launch_bounds__(256) void attn_score_kernel(
    const float* __restrict__ th, const int* __restrict__ amask,
    const uint32_t* __restrict__ w1bf, const float* __restrict__ b1,
    const float* __restrict__ w2, const float* __restrict__ b2,
    float* __restrict__ scores) {
    __shared__ uint32_t aLds[128 * 16];   // 128 rows x 32 k (bf16 packed)
    __shared__ float    rowSum[128];

    const int tid  = threadIdx.x;
    const int lane = tid & 31;
    const int wid  = tid >> 5;
    const int waveM = wid & 3;            // 4 waves along M (32 rows each)
    const int waveN = wid >> 2;           // 2 waves along N (64 cols each)
    const size_t rowBaseG = (size_t)blockIdx.x * 128;

    if (tid < 128) rowSum[tid] = 0.0f;
    __syncthreads();

    v8f zero = {};
    v8f acc[2][4];
#pragma unroll
    for (int mi = 0; mi < 2; ++mi)
#pragma unroll
        for (int ni = 0; ni < 4; ++ni) acc[mi][ni] = zero;

    for (int k0 = 0; k0 < BERT_DIM; k0 += 32) {
        // Cooperative load: 128 rows x 32 floats -> bf16 LDS tile
#pragma unroll
        for (int i = 0; i < 4; ++i) {
            int s  = tid + i * 256;       // 0..1023 float4 slots
            int r  = s >> 3;
            int c4 = s & 7;
            const float* src = th + (rowBaseG + r) * BERT_DIM + k0 + c4 * 4;
            const float4 v = *(const float4*)src;
            // prefetch the same slot of the next K slice (global_prefetch_b8)
            if (k0 + 32 < BERT_DIM) __builtin_prefetch(src + 32, 0, 3);
            aLds[r * 16 + c4 * 2]     = pack2(v.x, v.y);
            aLds[r * 16 + c4 * 2 + 1] = pack2(v.z, v.w);
        }
        __syncthreads();

        v16bf afr[2], bfr[4];
#pragma unroll
        for (int mi = 0; mi < 2; ++mi)
            afr[mi] = load_frag_a(aLds + (waveM * 32 + mi * 16) * 16, 16, lane);
#pragma unroll
        for (int ni = 0; ni < 4; ++ni)
            bfr[ni] = load_frag_b(w1bf, waveN * 64 + ni * 16, k0 >> 1, lane);
#pragma unroll
        for (int mi = 0; mi < 2; ++mi)
#pragma unroll
            for (int ni = 0; ni < 4; ++ni)
                acc[mi][ni] = __builtin_amdgcn_wmma_f32_16x16x32_bf16(
                    false, afr[mi], false, bfr[ni], (short)0, acc[mi][ni],
                    false, false);
        __syncthreads();
    }

    // Epilogue: tanh, dot with w2, cross-lane + cross-wave reduce per row.
    const float b2v = b2[0];
#pragma unroll
    for (int mi = 0; mi < 2; ++mi) {
        float s[8];
#pragma unroll
        for (int r = 0; r < 8; ++r) s[r] = 0.0f;
#pragma unroll
        for (int ni = 0; ni < 4; ++ni) {
            int col = waveN * 64 + ni * 16 + (lane & 15);
            float b1v = b1[col];
            float w2v = w2[col];
#pragma unroll
            for (int r = 0; r < 8; ++r)
                s[r] += tanhf(acc[mi][ni][r] + b1v) * w2v;
        }
#pragma unroll
        for (int r = 0; r < 8; ++r) {
            float v = s[r];
            v += __shfl_xor(v, 1, 32);
            v += __shfl_xor(v, 2, 32);
            v += __shfl_xor(v, 4, 32);
            v += __shfl_xor(v, 8, 32);    // reduce within each 16-lane half
            if ((lane & 15) == 0) {
                int row = waveM * 32 + mi * 16 + ((lane >> 4) ? 8 : 0) + r;
                atomicAdd(&rowSum[row], v);
            }
        }
    }
    __syncthreads();

    if (tid < 128) {
        size_t row = rowBaseG + tid;
        float sc = rowSum[tid] + b2v;
        if (amask[row] == 0) sc = -3.4028234663852886e38f;  // finfo(f32).min
        scores[row] = sc;
    }
}

// ===========================================================================
// Kernel 3: top-12 of masked scores per batch (ties -> lower index), indices
//           sorted ascending. One block per batch; one wave does the work.
//           (softmax is monotone and its weights are never consumed, so
//            top-k of masked scores == top-k of softmax weights)
// ===========================================================================
__global__ __launch_bounds__(256) void topk_kernel(
    const float* __restrict__ scores, int* __restrict__ topidx) {
    __shared__ float vals[SEQ];
    const int b = blockIdx.x, tid = threadIdx.x;
    vals[tid]       = scores[(size_t)b * SEQ + tid];
    vals[tid + 256] = scores[(size_t)b * SEQ + tid + 256];
    __syncthreads();

    if (tid < 32) {
        int chosen[N_COLS];
#pragma unroll 1
        for (int t = 0; t < N_COLS; ++t) {
            float bv = -INFINITY;
            int   bi = 0x7fffffff;
            for (int i = tid; i < SEQ; i += 32) {
                float v = vals[i];
                if (v > bv || (v == bv && i < bi)) { bv = v; bi = i; }
            }
#pragma unroll
            for (int off = 16; off > 0; off >>= 1) {
                float ov = __shfl_xor(bv, off, 32);
                int   oi = __shfl_xor(bi, off, 32);
                if (ov > bv || (ov == bv && oi < bi)) { bv = ov; bi = oi; }
            }
            chosen[t] = bi;          // all lanes agree
            vals[bi]  = -INFINITY;   // every lane writes same value/addr
        }
        if (tid == 0) {
            // insertion sort ascending (preserve token order)
#pragma unroll
            for (int i = 1; i < N_COLS; ++i) {
                int key = chosen[i], j = i - 1;
                while (j >= 0 && chosen[j] > key) { chosen[j + 1] = chosen[j]; --j; }
                chosen[j + 1] = key;
            }
#pragma unroll
            for (int i = 0; i < N_COLS; ++i) topidx[b * N_COLS + i] = chosen[i];
        }
    }
}

// ===========================================================================
// Kernel 4: gather 12 tokens, GELU(sel @ proj_w1^T + b1) @ proj_w2^T + b2,
//           transpose to [5,12], LayerNorm. One block per batch.
// ===========================================================================
__global__ __launch_bounds__(256) void project_kernel(
    const float* __restrict__ th, const uint32_t* __restrict__ pw1bf,
    const float* __restrict__ pb1, const float* __restrict__ pw2,
    const float* __restrict__ pb2, const float* __restrict__ gamma,
    const float* __restrict__ beta, const int* __restrict__ topidx,
    float* __restrict__ out) {
    __shared__ int      idxA[N_COLS];
    __shared__ uint32_t selLds[16 * 384];      // 16 rows x 768 bf16 (padded M)
    __shared__ float    fLds[N_COLS * 257];    // [12][256] +1 pad vs bank conflicts
    __shared__ float    outLds[N_ROWS * N_COLS];
    __shared__ float    stats[2];

    const int b = blockIdx.x, tid = threadIdx.x;
    if (tid < N_COLS) idxA[tid] = topidx[b * N_COLS + tid];
    __syncthreads();

    // Gather 12 selected rows, convert to bf16; zero pad rows 12..15.
    for (int s = tid; s < N_COLS * 384; s += 256) {
        int r = s / 384, c2 = s % 384;
        const float2 v = *(const float2*)(th + ((size_t)b * SEQ + idxA[r]) * BERT_DIM + c2 * 2);
        selLds[r * 384 + c2] = pack2(v.x, v.y);
    }
    for (int s = tid; s < 4 * 384; s += 256) selLds[N_COLS * 384 + s] = 0u;
    __syncthreads();

    // GEMM: [16 x 768] x [768 x 256], 8 waves each own 32 output cols.
    const int lane = tid & 31, wid = tid >> 5;
    const int n0 = wid * 32;
    v8f zero = {};
    v8f acc[2] = {zero, zero};
    for (int k0 = 0; k0 < BERT_DIM; k0 += 32) {
        v16bf a = load_frag_a(selLds + (k0 >> 1), 384, lane);
#pragma unroll
        for (int ni = 0; ni < 2; ++ni) {
            v16bf bf = load_frag_b(pw1bf, n0 + ni * 16, k0 >> 1, lane);
            acc[ni] = __builtin_amdgcn_wmma_f32_16x16x32_bf16(
                false, a, false, bf, (short)0, acc[ni], false, false);
        }
    }
    // Exact GELU, scatter into fLds[row][col]
#pragma unroll
    for (int ni = 0; ni < 2; ++ni) {
#pragma unroll
        for (int r = 0; r < 8; ++r) {
            int row = ((lane >> 4) ? 8 : 0) + r;
            int col = n0 + ni * 16 + (lane & 15);
            if (row < N_COLS) {
                float x = acc[ni][r] + pb1[col];
                fLds[row * 257 + col] = 0.5f * x * (1.0f + erff(x * 0.70710678118654752f));
            }
        }
    }
    __syncthreads();

    // Second projection (tiny): features[r][c] = sum_h f[c][h] * pw2[r][h] + pb2[r]
    if (tid < N_ROWS * N_COLS) {
        int r = tid / N_COLS, c = tid % N_COLS;
        float a = pb2[r];
        for (int h = 0; h < 256; ++h) a += fLds[c * 257 + h] * pw2[r * 256 + h];
        outLds[tid] = a;   // tid == r*12 + c
    }
    __syncthreads();

    if (tid == 0) {
        float m = 0.0f;
#pragma unroll
        for (int i = 0; i < N_ROWS * N_COLS; ++i) m += outLds[i];
        m *= (1.0f / (N_ROWS * N_COLS));
        float v = 0.0f;
#pragma unroll
        for (int i = 0; i < N_ROWS * N_COLS; ++i) {
            float d = outLds[i] - m;
            v += d * d;
        }
        v *= (1.0f / (N_ROWS * N_COLS));
        stats[0] = m;
        stats[1] = rsqrtf(v + 1e-5f);
    }
    __syncthreads();

    if (tid < N_ROWS * N_COLS)
        out[(size_t)b * (N_ROWS * N_COLS) + tid] =
            gamma[tid] * (outLds[tid] - stats[0]) * stats[1] + beta[tid];
}

// ===========================================================================
// Launch
// ===========================================================================
extern "C" void kernel_launch(void* const* d_in, const int* in_sizes, int n_in,
                              void* d_out, int out_size, void* d_ws, size_t ws_size,
                              hipStream_t stream) {
    const float* th      = (const float*)d_in[0];
    const int*   amask   = (const int*)d_in[1];
    const float* att_w1  = (const float*)d_in[2];
    const float* att_b1  = (const float*)d_in[3];
    const float* att_w2  = (const float*)d_in[4];
    const float* att_b2  = (const float*)d_in[5];
    const float* proj_w1 = (const float*)d_in[6];
    const float* proj_b1 = (const float*)d_in[7];
    const float* proj_w2 = (const float*)d_in[8];
    const float* proj_b2 = (const float*)d_in[9];
    const float* g       = (const float*)d_in[10];
    const float* be      = (const float*)d_in[11];
    float*       out     = (float*)d_out;

    char* ws = (char*)d_ws;
    uint16_t* w1bf   = (uint16_t*)(ws);                 // 128*768*2  = 196608 B
    uint16_t* pw1bf  = (uint16_t*)(ws + 196608);        // 256*768*2  = 393216 B
    float*    scores = (float*)(ws + 589824);           // 256*512*4  = 524288 B
    int*      topidx = (int*)(ws + 1114112);            // 256*12*4   =  12288 B

    // 1) weights -> bf16
    cvt_weights_kernel<<<(256 * BERT_DIM + 255) / 256, 256, 0, stream>>>(
        att_w1, proj_w1, w1bf, pw1bf);

    // 2) attention scores (WMMA bf16, memory-bound over 402 MB of activations)
    attn_score_kernel<<<(BATCH * SEQ) / 128, 256, 0, stream>>>(
        th, amask, (const uint32_t*)w1bf, att_b1, att_w2, att_b2, scores);

    // 3) top-12 token indices per batch
    topk_kernel<<<BATCH, 256, 0, stream>>>(scores, topidx);

    // 4) gather + projector (WMMA) + GELU + tiny GEMM + LayerNorm
    project_kernel<<<BATCH, 256, 0, stream>>>(
        th, (const uint32_t*)pw1bf, proj_b1, proj_w2, proj_b2, g, be, topidx, out);
}
